// VJEPAEncoder_18442589569030
// MI455X (gfx1250) — compile-verified, matrix operand order checked by reading, not run on hardware
//
#include <hip/hip_runtime.h>
#include <hip/hip_bf16.h>

// ---------------------------------------------------------------------------
// V-JEPA ViT encoder for gfx1250 (MI455X). bf16 WMMA GEMMs, f32 accumulate.
// B=32, T=2, C=3, HW=224, P=16, TS=2 -> tokens M = 32*196 = 6272
// D=768, DFF=3072, NH=12, DH=64, DEPTH=12
// ---------------------------------------------------------------------------

#define TOK_M   6272          // 49 * 128
#define DMODEL  768
#define DFF     3072
#define D3      2304
#define KCONV   1536
#define NTOK    196
#define NHEADS  12
#define DHEAD   64

typedef __bf16 v16bf __attribute__((ext_vector_type(16)));
typedef float  v8f   __attribute__((ext_vector_type(8)));
typedef int    v4i   __attribute__((ext_vector_type(4)));

union FragAB {
    v16bf bf;
    uint4 q[2];
};

__device__ __forceinline__ unsigned short f2bf(float f) {
    unsigned u = __float_as_uint(f);
    unsigned r = u + 0x7FFFu + ((u >> 16) & 1u);   // round-to-nearest-even
    return (unsigned short)(r >> 16);
}

// ---------------------------------------------------------------------------
// gfx1250 async global->LDS path (ASYNCcnt), guarded so compile never breaks.
// Prototype (from hipcc diagnostic): (int4 AS1*, int4 AS3*, imm, imm)
// ---------------------------------------------------------------------------
#if __has_builtin(__builtin_amdgcn_global_load_async_to_lds_b128) && \
    __has_builtin(__builtin_amdgcn_s_wait_asynccnt)
#define ASYNC_LDS 1
#else
#define ASYNC_LDS 0
#endif

#if ASYNC_LDS
__device__ __forceinline__ void async_copy16(const void* g, void* lds) {
    // Generic LDS address: low 32 bits are the LDS offset (ISA 10.2), so the
    // uintptr_t round-trip yields a valid address-space(3) pointer.
    __builtin_amdgcn_global_load_async_to_lds_b128(
        (__attribute__((address_space(1))) v4i*)(uintptr_t)g,
        (__attribute__((address_space(3))) v4i*)(unsigned int)(uintptr_t)lds,
        0, 0);
}
#endif

// ---------------------------------------------------------------------------
// RoPE tables: cos/sin[n][j], n in [0,196), j in [0,32)
// ---------------------------------------------------------------------------
__global__ void rope_init_kernel(float* __restrict__ rc, float* __restrict__ rs) {
    int idx = blockIdx.x * 256 + threadIdx.x;
    if (idx >= NTOK * 32) return;
    int n = idx >> 5, j = idx & 31;
    int row = n / 14, col = n % 14;
    int i = j & 15;
    float inv = powf(10000.f, -(float)i / 16.f);
    float ang = (j < 16 ? (float)row : (float)col) * inv;
    rc[idx] = cosf(ang);
    rs[idx] = sinf(ang);
}

// ---------------------------------------------------------------------------
// im2col: x (B,T,C,224,224) f32 -> Xcol (6272 x 1536) bf16
// k = ((c*2 + t)*16 + p)*16 + q
// ---------------------------------------------------------------------------
__global__ void im2col_kernel(const float* __restrict__ x,
                              unsigned short* __restrict__ xc) {
    long idx = (long)blockIdx.x * 256 + threadIdx.x;
    if (idx >= (long)TOK_M * KCONV) return;
    int  k = (int)(idx % KCONV);
    long m = idx / KCONV;
    int  b = (int)(m / NTOK);
    int  n = (int)(m % NTOK);
    int  H = n / 14, Wp = n % 14;
    int  q  = k & 15;
    int  pp = (k >> 4) & 15;
    int  t  = (k >> 8) & 1;
    int  c  = k >> 9;
    long src = ((((long)b * 2 + t) * 3 + c) * 224 + (H * 16 + pp)) * 224 + (Wp * 16 + q);
    xc[idx] = f2bf(x[src]);
}

// ---------------------------------------------------------------------------
// WMMA GEMM:  out(M x N) = A(M x K, bf16) * W(N x K, f32->bf16)^T  + bias
// grid = (N/128, M/128), block = 256 threads (8 waves), tile 128x128x32.
// EPI: 0 = store f32, 1 = residual add into f32, 2 = exact GELU -> bf16
// ---------------------------------------------------------------------------
template <int EPI>
__global__ __launch_bounds__(256) void gemm_wmma_kernel(
    const unsigned short* __restrict__ A,
    const float*          __restrict__ W,
    const float*          __restrict__ bias,
    float*                __restrict__ outF,
    unsigned short*       __restrict__ outBf,
    int K, int ldc)
{
    __shared__ __align__(16) unsigned short As[128 * 48];   // 12 KB, pad to kill conflicts
    __shared__ __align__(16) unsigned short Bs[128 * 48];   // 12 KB

    const int tid  = threadIdx.x;
    const int lane = tid & 31;
    const int wave = tid >> 5;
    const int wm   = wave >> 1;          // 0..3 -> M offset wm*32
    const int wn   = wave & 1;           // 0..1 -> N offset wn*64
    const long blockM = (long)blockIdx.y * 128;
    const long blockN = (long)blockIdx.x * 128;

    const int ldrow = tid >> 1;          // 0..127
    const int ldcb  = (tid & 1) * 16;    // 0 or 16

    v8f acc[2][4];
    #pragma unroll
    for (int i = 0; i < 2; i++)
        #pragma unroll
        for (int j = 0; j < 4; j++)
            #pragma unroll
            for (int e = 0; e < 8; e++) acc[i][j][e] = 0.f;

    const int mrow  = lane & 15;         // row within 16x16 frag
    const int khalf = (lane >> 4) * 8;   // K sub-base per ISA 16-bit A layout

    for (int kb = 0; kb < K; kb += 32) {
        // ---- stage A tile (bf16, 128x32) ----
#if ASYNC_LDS
        {
            const unsigned short* src = A + (blockM + ldrow) * (long)K + kb + ldcb;
            unsigned short* dst = &As[ldrow * 48 + ldcb];
            async_copy16(src,     dst);        // global_load_async_to_lds_b128
            async_copy16(src + 8, dst + 8);
        }
#else
        {
            const uint4* src = (const uint4*)(A + (blockM + ldrow) * (long)K + kb + ldcb);
            uint4 a0 = src[0], a1 = src[1];
            uint4* dst = (uint4*)&As[ldrow * 48 + ldcb];
            dst[0] = a0; dst[1] = a1;
        }
#endif
        // ---- stage W tile (f32 -> bf16, 128x32) ----
        {
            const float4* src = (const float4*)(W + (blockN + ldrow) * (long)K + kb + ldcb);
            float4 w0 = src[0], w1 = src[1], w2 = src[2], w3 = src[3];
            __align__(16) unsigned short t16[16] = {
                f2bf(w0.x), f2bf(w0.y), f2bf(w0.z), f2bf(w0.w),
                f2bf(w1.x), f2bf(w1.y), f2bf(w1.z), f2bf(w1.w),
                f2bf(w2.x), f2bf(w2.y), f2bf(w2.z), f2bf(w2.w),
                f2bf(w3.x), f2bf(w3.y), f2bf(w3.z), f2bf(w3.w)};
            uint4* dst = (uint4*)&Bs[ldrow * 48 + ldcb];
            dst[0] = *(const uint4*)&t16[0];
            dst[1] = *(const uint4*)&t16[8];
        }
        if (kb + 32 < K) {   // prefetch next tiles -> global_prefetch_b8
            __builtin_prefetch(A + (blockM + ldrow) * (long)K + kb + 32 + ldcb, 0, 0);
            __builtin_prefetch(W + (blockN + ldrow) * (long)K + kb + 32 + ldcb, 0, 0);
        }
#if ASYNC_LDS
        __builtin_amdgcn_s_wait_asynccnt(0);   // own async copies landed in LDS
#endif
        __syncthreads();

        // ---- load fragments (ISA 16-bit 16x32 layout) ----
        FragAB fa[2], fb[4];
        #pragma unroll
        for (int mi = 0; mi < 2; mi++) {
            const unsigned short* p = &As[(wm * 32 + mi * 16 + mrow) * 48 + khalf];
            fa[mi].q[0] = *(const uint4*)(p);
            fa[mi].q[1] = *(const uint4*)(p + 16);
        }
        #pragma unroll
        for (int ni = 0; ni < 4; ni++) {
            const unsigned short* p = &Bs[(wn * 64 + ni * 16 + mrow) * 48 + khalf];
            fb[ni].q[0] = *(const uint4*)(p);
            fb[ni].q[1] = *(const uint4*)(p + 16);
        }

        #pragma unroll
        for (int mi = 0; mi < 2; mi++)
            #pragma unroll
            for (int ni = 0; ni < 4; ni++)
                acc[mi][ni] = __builtin_amdgcn_wmma_f32_16x16x32_bf16(
                    false, fa[mi].bf, false, fb[ni].bf,
                    (short)0, acc[mi][ni], false, false);
        __syncthreads();
    }

    // ---- epilogue ----
    #pragma unroll
    for (int ni = 0; ni < 4; ni++) {
        const long col = blockN + wn * 64 + ni * 16 + (lane & 15);
        const float bv = bias[col];
        #pragma unroll
        for (int mi = 0; mi < 2; mi++) {
            #pragma unroll
            for (int r = 0; r < 8; r++) {
                const long row = blockM + wm * 32 + mi * 16 + r + 8 * (lane >> 4);
                const long idx = row * (long)ldc + col;
                float v = acc[mi][ni][r] + bv;
                if (EPI == 0) {
                    outF[idx] = v;
                } else if (EPI == 1) {
                    outF[idx] += v;
                } else {
                    float g = 0.5f * v * (1.f + erff(v * 0.70710678118654752f));
                    outBf[idx] = f2bf(g);
                }
            }
        }
    }
}

// ---------------------------------------------------------------------------
// LayerNorm over D=768, one block per row. BF16OUT: write bf16 else f32.
// ---------------------------------------------------------------------------
template <bool BF16OUT>
__global__ __launch_bounds__(256) void layernorm_kernel(
    const float* __restrict__ X, const float* __restrict__ g,
    const float* __restrict__ bta,
    unsigned short* __restrict__ outBf, float* __restrict__ outF)
{
    __shared__ float s1[256], s2[256];
    const int row = blockIdx.x, tid = threadIdx.x;
    const float* xr = X + (long)row * DMODEL;
    float sum = 0.f, sq = 0.f;
    for (int c = tid; c < DMODEL; c += 256) { float v = xr[c]; sum += v; sq += v * v; }
    s1[tid] = sum; s2[tid] = sq; __syncthreads();
    for (int off = 128; off > 0; off >>= 1) {
        if (tid < off) { s1[tid] += s1[tid + off]; s2[tid] += s2[tid + off]; }
        __syncthreads();
    }
    const float mean = s1[0] * (1.f / DMODEL);
    const float var  = s2[0] * (1.f / DMODEL) - mean * mean;
    const float rstd = rsqrtf(var + 1e-5f);
    for (int c = tid; c < DMODEL; c += 256) {
        float y = (xr[c] - mean) * rstd * g[c] + bta[c];
        if (BF16OUT) outBf[(long)row * DMODEL + c] = f2bf(y);
        else         outF [(long)row * DMODEL + c] = y;
    }
}

// ---------------------------------------------------------------------------
// Attention (+RoPE) per (b,h). K rows RoPE'd into LDS; per-thread row with
// two-pass softmax; V streamed from global (broadcast reads). Writes bf16 o.
// ---------------------------------------------------------------------------
__global__ __launch_bounds__(256) void attention_kernel(
    const float* __restrict__ qkv, const float* __restrict__ rc,
    const float* __restrict__ rs,  unsigned short* __restrict__ obf)
{
    __shared__ float kR[NTOK * DHEAD];   // 50 KB
    const int bh = blockIdx.x;
    const int b  = bh / NHEADS, h = bh % NHEADS;
    const int tid = threadIdx.x;
    const long rowBase = (long)b * NTOK;

    for (int idx = tid; idx < NTOK * 32; idx += 256) {
        int n = idx >> 5, j = idx & 31;
        long base = (rowBase + n) * (long)D3 + DMODEL + h * DHEAD;
        float xe = qkv[base + 2 * j], xo = qkv[base + 2 * j + 1];
        float c = rc[n * 32 + j], s = rs[n * 32 + j];
        kR[n * DHEAD + 2 * j]     = xe * c - xo * s;
        kR[n * DHEAD + 2 * j + 1] = xe * s + xo * c;
    }
    __syncthreads();

    if (tid < NTOK) {
        const int n = tid;
        float qv[DHEAD];
        long base = (rowBase + n) * (long)D3 + h * DHEAD;
        #pragma unroll
        for (int j = 0; j < 32; j++) {
            float xe = qkv[base + 2 * j], xo = qkv[base + 2 * j + 1];
            float c = rc[n * 32 + j], s = rs[n * 32 + j];
            qv[2 * j]     = xe * c - xo * s;
            qv[2 * j + 1] = xe * s + xo * c;
        }
        float mx = -3.0e38f;
        for (int m = 0; m < NTOK; m++) {
            float sd = 0.f;
            #pragma unroll
            for (int d = 0; d < DHEAD; d++) sd += qv[d] * kR[m * DHEAD + d];
            mx = fmaxf(mx, sd * 0.125f);
        }
        float o[DHEAD];
        #pragma unroll
        for (int d = 0; d < DHEAD; d++) o[d] = 0.f;
        float sum = 0.f;
        const float* vbase = qkv + rowBase * (long)D3 + 2 * DMODEL + h * DHEAD;
        for (int m = 0; m < NTOK; m++) {
            float sd = 0.f;
            #pragma unroll
            for (int d = 0; d < DHEAD; d++) sd += qv[d] * kR[m * DHEAD + d];
            float p = __expf(sd * 0.125f - mx);
            sum += p;
            const float* vr = vbase + (long)m * D3;
            #pragma unroll
            for (int d = 0; d < DHEAD; d++) o[d] += p * vr[d];
        }
        float inv = 1.f / sum;
        long ob = (rowBase + n) * (long)DMODEL + h * DHEAD;
        #pragma unroll
        for (int d = 0; d < DHEAD; d++) obf[ob + d] = f2bf(o[d] * inv);
    }
}

// ---------------------------------------------------------------------------
// Host launcher
// ---------------------------------------------------------------------------
extern "C" void kernel_launch(void* const* d_in, const int* in_sizes, int n_in,
                              void* d_out, int out_size, void* d_ws, size_t ws_size,
                              hipStream_t stream) {
    const float* x      = (const float*)d_in[0];
    const float* conv_w = (const float*)d_in[1];
    const float* conv_b = (const float*)d_in[2];
    const float* ln1_g  = (const float*)d_in[3];
    const float* ln1_b  = (const float*)d_in[4];
    const float* qkv_w  = (const float*)d_in[5];
    const float* qkv_b  = (const float*)d_in[6];
    const float* proj_w = (const float*)d_in[7];
    const float* proj_b = (const float*)d_in[8];
    const float* ln2_g  = (const float*)d_in[9];
    const float* ln2_b  = (const float*)d_in[10];
    const float* mlp_w1 = (const float*)d_in[11];
    const float* mlp_b1 = (const float*)d_in[12];
    const float* mlp_w2 = (const float*)d_in[13];
    const float* mlp_w2b= (const float*)d_in[14];
    const float* lnf_g  = (const float*)d_in[15];
    const float* lnf_b  = (const float*)d_in[16];

    // workspace carve (256B aligned)
    char* p = (char*)d_ws;
    auto carve = [&](size_t bytes) -> void* {
        void* r = (void*)p;
        p += (bytes + 255) & ~(size_t)255;
        return r;
    };
    float* ropeC  = (float*)carve((size_t)NTOK * 32 * 4);
    float* ropeS  = (float*)carve((size_t)NTOK * 32 * 4);
    float* tokens = (float*)carve((size_t)TOK_M * DMODEL * 4);        // 19.3 MB
    float* qkvbuf = (float*)carve((size_t)TOK_M * D3 * 4);            // 57.8 MB
    unsigned short* hb = (unsigned short*)carve((size_t)TOK_M * DFF * 2); // 38.5 MB
    unsigned short* h2b  = (unsigned short*)qkvbuf;  // alias: qkv dead after attention
    unsigned short* xcol = hb;                       // alias: im2col before layer 0

    rope_init_kernel<<<(NTOK * 32 + 255) / 256, 256, 0, stream>>>(ropeC, ropeS);

    // patch embed
    im2col_kernel<<<(int)(((long)TOK_M * KCONV + 255) / 256), 256, 0, stream>>>(x, xcol);
    gemm_wmma_kernel<0><<<dim3(DMODEL / 128, TOK_M / 128), 256, 0, stream>>>(
        xcol, conv_w, conv_b, tokens, nullptr, KCONV, DMODEL);

    for (int l = 0; l < 12; l++) {
        const float* wqkv = qkv_w  + (size_t)l * D3 * DMODEL;
        const float* bqkv = qkv_b  + (size_t)l * D3;
        const float* wprj = proj_w + (size_t)l * DMODEL * DMODEL;
        const float* bprj = proj_b + (size_t)l * DMODEL;
        const float* w1   = mlp_w1 + (size_t)l * DFF * DMODEL;
        const float* b1   = mlp_b1 + (size_t)l * DFF;
        const float* w2   = mlp_w2 + (size_t)l * DMODEL * DFF;
        const float* b2   = mlp_w2b+ (size_t)l * DMODEL;

        layernorm_kernel<true><<<TOK_M, 256, 0, stream>>>(
            tokens, ln1_g + l * DMODEL, ln1_b + l * DMODEL, hb, nullptr);

        gemm_wmma_kernel<0><<<dim3(D3 / 128, TOK_M / 128), 256, 0, stream>>>(
            hb, wqkv, bqkv, qkvbuf, nullptr, DMODEL, D3);

        attention_kernel<<<32 * NHEADS, 256, 0, stream>>>(qkvbuf, ropeC, ropeS, hb);

        gemm_wmma_kernel<1><<<dim3(DMODEL / 128, TOK_M / 128), 256, 0, stream>>>(
            hb, wprj, bprj, tokens, nullptr, DMODEL, DMODEL);

        layernorm_kernel<true><<<TOK_M, 256, 0, stream>>>(
            tokens, ln2_g + l * DMODEL, ln2_b + l * DMODEL, hb, nullptr);

        gemm_wmma_kernel<2><<<dim3(DFF / 128, TOK_M / 128), 256, 0, stream>>>(
            hb, w1, b1, nullptr, h2b, DMODEL, DFF);

        gemm_wmma_kernel<1><<<dim3(DMODEL / 128, TOK_M / 128), 256, 0, stream>>>(
            h2b, w2, b2, tokens, nullptr, DFF, DMODEL);
    }

    layernorm_kernel<false><<<TOK_M, 256, 0, stream>>>(
        tokens, lnf_g, lnf_b, nullptr, (float*)d_out);
}
